// PointRend_17952963297881
// MI455X (gfx1250) — matrix operand order repaired and checked by reading, not compiled, not run on hardware
//
#include <hip/hip_runtime.h>
#include <math.h>

typedef __attribute__((ext_vector_type(16))) __bf16 v16bf;
typedef __attribute__((ext_vector_type(8)))  __bf16 v8bf;
typedef __attribute__((ext_vector_type(8)))  float  v8f;
typedef __attribute__((ext_vector_type(4)))  int    v4i;

#define NPTS 131072          // B * P = 2 * 65536
#define DHW  2097152         // 128^3
#define HW   16384           // 128^2

#if defined(__has_builtin)
# if __has_builtin(__builtin_amdgcn_global_load_async_to_lds_b128)
#  define HAVE_ASYNC_LDS 1
# endif
#endif

// ---------------------------------------------------------------------------
// Gather + positional encoding -> X0 (15 x N) bf16, channel-major
// ---------------------------------------------------------------------------
__global__ __launch_bounds__(256) void gather_kernel(
    const float* __restrict__ pre, const float* __restrict__ fmap,
    const int* __restrict__ g, __bf16* __restrict__ X)
{
  int n = blockIdx.x * 256 + threadIdx.x;      // 0..NPTS-1
  int b = n >> 16;
  const int* gp = g + (size_t)n * 3;
  int z = gp[0], y = gp[1], x = gp[2];
  float fz = (float)z, fy = (float)y, fx = (float)x;
  z = z < 0 ? 0 : (z > 127 ? 127 : z);
  y = y < 0 ? 0 : (y > 127 ? 127 : y);
  x = x < 0 ? 0 : (x > 127 ? 127 : x);
  size_t vox = (size_t)z * HW + y * 128 + x;
  X[0 * NPTS + n] = (__bf16)pre[(size_t)b * DHW + vox];
  X[1 * NPTS + n] = (__bf16)sinf(fz);
  X[2 * NPTS + n] = (__bf16)sinf(fy);
  X[3 * NPTS + n] = (__bf16)sinf(fx);
  X[4 * NPTS + n] = (__bf16)cosf(fz);
  X[5 * NPTS + n] = (__bf16)cosf(fy);
  X[6 * NPTS + n] = (__bf16)cosf(fx);
#pragma unroll
  for (int c = 0; c < 8; ++c)
    X[(size_t)(7 + c) * NPTS + n] = (__bf16)fmap[((size_t)(b * 8 + c)) * DHW + vox];
}

// ---------------------------------------------------------------------------
// WMMA GEMM: Y(O x N) = W(O x C, fp32) @ X(C x N, bf16), raw bf16 output.
// Block tile 64(O) x 128(N), 8 waves; each wave owns a 16-row stripe and a
// 64-column span = four 16x16 accumulators fed by one shared A fragment.
// GUARD=true only for C=15: branchless (clamp index + select-zero), so the
// K-remainder costs v_cndmask, not exec-mask control flow.
// ---------------------------------------------------------------------------
template <bool GUARD>
__global__ __launch_bounds__(256) void gemm_wmma(
    const __bf16* __restrict__ X, const float* __restrict__ W,
    __bf16* __restrict__ Y, int C, int O)
{
  __shared__ __align__(16) __bf16 Alds[64 * 32];    // [m][k], 64B rows
  __shared__ __align__(16) __bf16 Blds[128 * 40];   // [n][k], 80B rows (padded)
  const int tid  = threadIdx.x;
  const int lane = tid & 31;
  const int wave = tid >> 5;
  const int n0 = blockIdx.x * 128;
  const int o0 = blockIdx.y * 64;
  const int mw = (wave & 3) << 4;    // wave's M offset in tile
  const int nw = (wave >> 2) << 6;   // wave's N offset (64 cols per wave)

  const v8f vzero = {0.f,0.f,0.f,0.f,0.f,0.f,0.f,0.f};
  v8f acc[4] = {vzero, vzero, vzero, vzero};

  const int ksteps = (C + 31) >> 5;
  for (int ks = 0; ks < ksteps; ++ks) {
    const int k0 = ks << 5;

    // ---- stage A: 64x32 W tile, fp32 -> bf16 --------------------------
    {
      int mm = tid & 63;
      int k8 = (tid >> 6) << 3;            // 0,8,16,24
      if (!GUARD) {
        const float* wrow = &W[(size_t)(o0 + mm) * C + k0 + k8];
        float4 f0 = *(const float4*)wrow;
        float4 f1 = *(const float4*)(wrow + 4);
        v8bf av;
        av[0] = (__bf16)f0.x; av[1] = (__bf16)f0.y;
        av[2] = (__bf16)f0.z; av[3] = (__bf16)f0.w;
        av[4] = (__bf16)f1.x; av[5] = (__bf16)f1.y;
        av[6] = (__bf16)f1.z; av[7] = (__bf16)f1.w;
        *(v8bf*)&Alds[mm * 32 + k8] = av;
      } else {
        // branchless: clamp K into [0, C-1] (always a legal address inside
        // row mm), then select 0 for the zero-padded tail.
        const float* wrow = &W[(size_t)(o0 + mm) * C];
        v8bf av;
#pragma unroll
        for (int j = 0; j < 8; ++j) {
          int cc = k0 + k8 + j;
          int ccl = cc < C ? cc : C - 1;
          float wv = wrow[ccl];
          av[j] = (__bf16)(cc < C ? wv : 0.f);
        }
        *(v8bf*)&Alds[mm * 32 + k8] = av;
      }
    }
    // ---- stage B: 32(k) x 128(n) X tile, transposed into [n][k] -------
    {
      int kk = tid >> 3;                   // 0..31
      int ns = (tid & 7) << 4;             // 0,16,...,112
      int cc = k0 + kk;
      int ccl = (!GUARD || cc < C) ? cc : C - 1;   // clamped row (legal)
      const __bf16* xr = &X[(size_t)ccl * NPTS + n0 + ns];
      v8bf x0 = *(const v8bf*)xr;
      v8bf x1 = *(const v8bf*)(xr + 8);
      if (GUARD && cc >= C) {
#pragma unroll
        for (int j = 0; j < 8; ++j) { x0[j] = (__bf16)0.f; x1[j] = (__bf16)0.f; }
      }
#pragma unroll
      for (int j = 0; j < 8; ++j) {
        Blds[(ns + j) * 40 + kk]     = x0[j];
        Blds[(ns + 8 + j) * 40 + kk] = x1[j];
      }
    }
    __syncthreads();

    // A fragment (16x32 16-bit layout, ISA 7.12.2):
    // lane = M + 16*((K%16)/8); element e = (K%8) + 8*(K/16)
    int Am = mw + (lane & 15);
    int kh = (lane >> 4) << 3;                       // 0 or 8
    v8bf alo = *(const v8bf*)&Alds[Am * 32 + kh];        // K = kh..kh+7
    v8bf ahi = *(const v8bf*)&Alds[Am * 32 + kh + 16];   // K = kh+16..kh+23
    v16bf af = __builtin_shufflevector(alo, ahi,
        0,1,2,3,4,5,6,7,8,9,10,11,12,13,14,15);

    // B fragments (32x16): lane = Ncol + 16*((K%16)/8); element e = (K%8) + 8*(K/16)
    int bk = (lane >> 4) << 3;                       // 0 or 8
#pragma unroll
    for (int t = 0; t < 4; ++t) {
      int bn = nw + (t << 4) + (lane & 15);
      v8bf bl = *(const v8bf*)&Blds[bn * 40 + bk];
      v8bf bh = *(const v8bf*)&Blds[bn * 40 + bk + 16];
      v16bf bf = __builtin_shufflevector(bl, bh,
          0,1,2,3,4,5,6,7,8,9,10,11,12,13,14,15);
      acc[t] = __builtin_amdgcn_wmma_f32_16x16x32_bf16(false, af, false, bf,
                                                       (short)0, acc[t], false, false);
    }
    __syncthreads();
  }

  // D layout (f32 16x16): lane: Ncol = lane%16; VGPR r: M = r + 8*(lane/16)
  int rowBase = o0 + mw + ((lane >> 4) << 3);
#pragma unroll
  for (int t = 0; t < 4; ++t) {
    int col = n0 + nw + (t << 4) + (lane & 15);
#pragma unroll
    for (int r = 0; r < 8; ++r)
      Y[(size_t)(rowBase + r) * NPTS + col] = (__bf16)acc[t][r];
  }
}

// ---------------------------------------------------------------------------
// Per-channel batch stats: mean and rsqrt(var + eps)
// ---------------------------------------------------------------------------
__global__ __launch_bounds__(256) void stats_kernel(
    const __bf16* __restrict__ Y, float2* __restrict__ st)
{
  int o = blockIdx.x, tid = threadIdx.x;
  const __bf16* yp = Y + (size_t)o * NPTS;
  float s = 0.f, q = 0.f;
  for (int i = tid * 8; i < NPTS; i += 256 * 8) {
    v8bf v = *(const v8bf*)&yp[i];
#pragma unroll
    for (int j = 0; j < 8; ++j) {
      float f = (float)v[j];
      s += f; q += f * f;
    }
  }
  __shared__ float ss[256], sq[256];
  ss[tid] = s; sq[tid] = q;
  __syncthreads();
  for (int w = 128; w > 0; w >>= 1) {
    if (tid < w) { ss[tid] += ss[tid + w]; sq[tid] += sq[tid + w]; }
    __syncthreads();
  }
  if (tid == 0) {
    float m = ss[0] * (1.f / NPTS);
    float var = sq[0] * (1.f / NPTS) - m * m;
    st[o] = make_float2(m, rsqrtf(var + 1e-5f));
  }
}

// ---------------------------------------------------------------------------
// In-place BN + leaky ReLU (0.01), 8 elements per thread (v8bf)
// ---------------------------------------------------------------------------
__global__ __launch_bounds__(256) void bnlrelu_kernel(
    __bf16* __restrict__ Y, const float2* __restrict__ st,
    const float* __restrict__ gm, const float* __restrict__ bt)
{
  size_t idx = ((size_t)blockIdx.x * 256 + threadIdx.x) * 8;
  int o = (int)(idx >> 17);                 // N = 2^17, all 8 share channel
  float2 s = st[o];
  float gmo = gm[o], bto = bt[o];
  v8bf v = *(const v8bf*)&Y[idx];
#pragma unroll
  for (int j = 0; j < 8; ++j) {
    float f = gmo * ((float)v[j] - s.x) * s.y + bto;
    v[j] = (__bf16)(f > 0.f ? f : 0.01f * f);
  }
  *(v8bf*)&Y[idx] = v;
}

// ---------------------------------------------------------------------------
// BN + lrelu of Y2 plus residual projection -> next block input (bf16)
// ---------------------------------------------------------------------------
__global__ __launch_bounds__(256) void combine_kernel(
    const __bf16* __restrict__ Y2, const __bf16* __restrict__ Pj,
    const float2* __restrict__ st, const float* __restrict__ gm,
    const float* __restrict__ bt, __bf16* __restrict__ Xout)
{
  size_t idx = ((size_t)blockIdx.x * 256 + threadIdx.x) * 8;
  int o = (int)(idx >> 17);
  float2 s = st[o];
  float gmo = gm[o], bto = bt[o];
  v8bf v = *(const v8bf*)&Y2[idx];
  v8bf p = *(const v8bf*)&Pj[idx];
#pragma unroll
  for (int j = 0; j < 8; ++j) {
    float f = gmo * ((float)v[j] - s.x) * s.y + bto;
    f = (f > 0.f ? f : 0.01f * f) + (float)p[j];
    v[j] = (__bf16)f;
  }
  *(v8bf*)&Xout[idx] = v;
}

// ---------------------------------------------------------------------------
// Final 128 -> 1 layer: out[n] = pb + sum_c pw[c] * X[c,n]
// ---------------------------------------------------------------------------
__global__ __launch_bounds__(256) void pwlayer_kernel(
    const __bf16* __restrict__ X, const float* __restrict__ pwv,
    const float* __restrict__ pbv, float* __restrict__ outv)
{
  int n = blockIdx.x * 256 + threadIdx.x;
  float s = pbv[0];
#pragma unroll 8
  for (int c = 0; c < 128; ++c)
    s += pwv[c] * (float)X[(size_t)c * NPTS + n];
  outv[n] = s;
}

// ---------------------------------------------------------------------------
// Volume copy (pre_seg -> vol) via the gfx1250 async global->LDS path
// (ASYNCcnt-tracked); falls back to a plain vectorized copy if unavailable.
// ---------------------------------------------------------------------------
__global__ __launch_bounds__(256) void copy_vol_kernel(
    const float4* __restrict__ src, float4* __restrict__ dst, int n4)
{
  int i = blockIdx.x * 256 + threadIdx.x;
#ifdef HAVE_ASYNC_LDS
  __shared__ __align__(16) float4 buf[256];
  if (i < n4) {
    __builtin_amdgcn_global_load_async_to_lds_b128(
        (__attribute__((address_space(1))) v4i*)(src + i),
        (__attribute__((address_space(3))) v4i*)(&buf[threadIdx.x]),
        0, 0);
#if defined(__has_builtin) && __has_builtin(__builtin_amdgcn_s_wait_asynccnt)
    __builtin_amdgcn_s_wait_asynccnt(0);
#else
    asm volatile("s_wait_asynccnt 0" ::: "memory");
#endif
    dst[i] = buf[threadIdx.x];
  }
#else
  if (i < n4) dst[i] = src[i];
#endif
}

// ---------------------------------------------------------------------------
// Scatter refined point predictions into the volume
// ---------------------------------------------------------------------------
__global__ __launch_bounds__(256) void scatter_kernel(
    const float* __restrict__ outv, const int* __restrict__ g,
    float* __restrict__ vol)
{
  int n = blockIdx.x * 256 + threadIdx.x;
  int b = n >> 16;
  const int* gp = g + (size_t)n * 3;
  int z = gp[0], y = gp[1], x = gp[2];
  z = z < 0 ? 0 : (z > 127 ? 127 : z);
  y = y < 0 ? 0 : (y > 127 ? 127 : y);
  x = x < 0 ? 0 : (x > 127 ? 127 : x);
  vol[(size_t)b * DHW + (size_t)z * HW + y * 128 + x] = outv[n];
}

// ---------------------------------------------------------------------------
// Direct 3x3x3 conv, SAME zero padding. ACT: 0 = relu, 1 = sigmoid
// ---------------------------------------------------------------------------
template <int CI, int CO, int ACT>
__global__ __launch_bounds__(256) void conv3d_kernel(
    const float* __restrict__ in, const float* __restrict__ w,
    const float* __restrict__ bias, float* __restrict__ out)
{
  int idx = blockIdx.x * 256 + threadIdx.x;        // over B*D*H*W
  int bb = idx / DHW;
  int rem = idx - bb * DHW;
  int z = rem / HW;
  int rr = rem - z * HW;
  int y = rr >> 7;
  int x = rr & 127;

  float acc[CO];
#pragma unroll
  for (int oc = 0; oc < CO; ++oc) acc[oc] = bias[oc];

#pragma unroll
  for (int ic = 0; ic < CI; ++ic) {
    const float* ip = in + ((size_t)(bb * CI + ic)) * DHW;
#pragma unroll
    for (int dz = -1; dz <= 1; ++dz) {
      int zz = z + dz; if ((unsigned)zz >= 128u) continue;
#pragma unroll
      for (int dy = -1; dy <= 1; ++dy) {
        int yy = y + dy; if ((unsigned)yy >= 128u) continue;
#pragma unroll
        for (int dx = -1; dx <= 1; ++dx) {
          int xx = x + dx; if ((unsigned)xx >= 128u) continue;
          float v = ip[(size_t)zz * HW + yy * 128 + xx];
          int t = ((dz + 1) * 3 + (dy + 1)) * 3 + (dx + 1);
#pragma unroll
          for (int oc = 0; oc < CO; ++oc)
            acc[oc] += w[(oc * CI + ic) * 27 + t] * v;
        }
      }
    }
  }
#pragma unroll
  for (int oc = 0; oc < CO; ++oc) {
    float v = acc[oc];
    if (ACT == 0) v = v > 0.f ? v : 0.f;
    else          v = 1.f / (1.f + __expf(-v));
    out[((size_t)(bb * CO + oc)) * DHW + rem] = v;
  }
}

// ---------------------------------------------------------------------------
extern "C" void kernel_launch(void* const* d_in, const int* in_sizes, int n_in,
                              void* d_out, int out_size, void* d_ws, size_t ws_size,
                              hipStream_t stream)
{
  const float* pre_seg = (const float*)d_in[0];
  const float* fmap    = (const float*)d_in[1];
  const int*   g       = (const int*)d_in[2];
  const float* w[6]; const float* gm[6]; const float* bt[6];
  for (int i = 0; i < 6; ++i) {
    w[i]  = (const float*)d_in[3 + 3 * i];
    gm[i] = (const float*)d_in[4 + 3 * i];
    bt[i] = (const float*)d_in[5 + 3 * i];
  }
  const float* pj[3] = { (const float*)d_in[21], (const float*)d_in[22],
                         (const float*)d_in[23] };
  const float* pwv = (const float*)d_in[24];
  const float* pbv = (const float*)d_in[25];
  const float* cw[3] = { (const float*)d_in[26], (const float*)d_in[28],
                         (const float*)d_in[30] };
  const float* cb[3] = { (const float*)d_in[27], (const float*)d_in[29],
                         (const float*)d_in[31] };

  char* ws = (char*)d_ws;
  const size_t SLAB = (size_t)512 * NPTS * 2;       // 128 MiB bf16 slab
  __bf16* A  = (__bf16*)ws;                         // current block input (r)
  __bf16* Bm = (__bf16*)(ws + SLAB);                // mid / proj
  __bf16* Cm = (__bf16*)(ws + 2 * SLAB);            // second conv output
  float2* st  = (float2*)(ws + 3 * SLAB);           // per-channel stats
  float*  outv = (float*)(ws + 3 * SLAB + 8192);    // (N) refined values
  float*  vol = (float*)Bm;                         // 16.8 MB volume
  float*  h1  = (float*)Cm;                         // 67 MB (4ch)
  float*  h2  = (float*)(ws + SLAB + (size_t)32 * 1024 * 1024);

  const dim3 thr(256);
  const int CH[7] = {15, 64, 128, 256, 512, 256, 128};

  auto launch_gemm = [&](const __bf16* Xp, const float* Wp, __bf16* Yp,
                         int Cc, int Oo) {
    dim3 grid(NPTS / 128, Oo / 64);
    if (Cc & 31)
      gemm_wmma<true><<<grid, thr, 0, stream>>>(Xp, Wp, Yp, Cc, Oo);
    else
      gemm_wmma<false><<<grid, thr, 0, stream>>>(Xp, Wp, Yp, Cc, Oo);
  };

  gather_kernel<<<NPTS / 256, thr, 0, stream>>>(pre_seg, fmap, g, A);

  for (int p = 0; p < 3; ++p) {
    int cin = CH[2 * p], mid = CH[2 * p + 1], cout = CH[2 * p + 2];
    launch_gemm(A, w[2 * p], Bm, cin, mid);
    stats_kernel<<<mid, thr, 0, stream>>>(Bm, st);
    bnlrelu_kernel<<<mid * (NPTS / 2048), thr, 0, stream>>>(Bm, st, gm[2 * p], bt[2 * p]);
    launch_gemm(Bm, w[2 * p + 1], Cm, mid, cout);
    stats_kernel<<<cout, thr, 0, stream>>>(Cm, st);
    launch_gemm(A, pj[p], Bm, cin, cout);
    combine_kernel<<<cout * (NPTS / 2048), thr, 0, stream>>>(Cm, Bm, st,
                                                             gm[2 * p + 1], bt[2 * p + 1], A);
  }

  pwlayer_kernel<<<NPTS / 256, thr, 0, stream>>>(A, pwv, pbv, outv);

  copy_vol_kernel<<<(2 * DHW / 4) / 256, thr, 0, stream>>>(
      (const float4*)pre_seg, (float4*)vol, 2 * DHW / 4);
  scatter_kernel<<<NPTS / 256, thr, 0, stream>>>(outv, g, vol);

  conv3d_kernel<1, 4, 0><<<(2 * DHW) / 256, thr, 0, stream>>>(vol, cw[0], cb[0], h1);
  conv3d_kernel<4, 4, 0><<<(2 * DHW) / 256, thr, 0, stream>>>(h1, cw[1], cb[1], h2);
  conv3d_kernel<4, 1, 1><<<(2 * DHW) / 256, thr, 0, stream>>>(h2, cw[2], cb[2], (float*)d_out);
}
